// Int4LinearWeightOnly_74191265071656
// MI455X (gfx1250) — matrix operand order repaired and checked by reading, not compiled, not run on hardware
//
#include <hip/hip_runtime.h>

// ---- CDNA5 WMMA vector types --------------------------------------------
typedef _Float16 v16h __attribute__((ext_vector_type(16)));
typedef _Float16 v8h  __attribute__((ext_vector_type(8)));
typedef _Float16 v4h  __attribute__((ext_vector_type(4)));
typedef _Float16 v2h  __attribute__((ext_vector_type(2)));
typedef float    v8f  __attribute__((ext_vector_type(8)));

#define BM   128   // block tile M
#define BN   128   // block tile N
#define BK   64    // block tile K (two WMMA k-steps of 32)
#define LDSK 72    // padded K stride in LDS (breaks bank alignment, keeps 16B align)

// out[m,n] = sum_k x[m,k] * int4(w[n,k]) * scale[n] + bias[n]
// f16 WMMA with f32 accumulation; scale/bias applied in f32 epilogue.
__global__ __launch_bounds__(256)
void int4linear_wmma_f16(const float* __restrict__ x,     // [M,K] fp32
                         const int*   __restrict__ wp,    // [N,K/2] int32, one byte (2 nibbles) each
                         const float* __restrict__ scale, // [N]
                         const float* __restrict__ bias,  // [N]
                         float* __restrict__ out,         // [M,N]
                         int M, int N, int K, int Kh)     // Kh = K/2
{
    __shared__ __align__(16) _Float16 As[BM][LDSK];  // x tile,  [M][K] f16
    __shared__ __align__(16) _Float16 Bs[BN][LDSK];  // w tile,  [N][K] f16 (int4 values, exact)

    const int tid  = threadIdx.x;
    const int lane = tid & 31;
    const int wid  = tid >> 5;
    const int wm   = (wid >> 2) * 64;  // wave M offset (2 waves in M)
    const int wn   = (wid & 3)  * 32;  // wave N offset (4 waves in N)

    const int n0 = blockIdx.x * BN;
    const int m0 = blockIdx.y * BM;

    // --- per-thread staging geometry (fixed across K tiles) ---
    // x tile: 128 rows x 16 float4 cols = 2048 float4, 8/thread (row += 16 per i)
    const int xrow = tid >> 4;          // 0..15
    const int xcol = (tid & 15) * 4;    // element column of float4
    // w tile: 128 rows x 8 int4 cols = 1024 int4, 4/thread (row += 32 per i)
    const int wrow = tid >> 3;          // 0..31
    const int wj4  = (tid & 7) * 4;     // int index of int4 within row slice

    const float* xbase = x  + (size_t)(m0 + xrow) * K  + xcol;
    const int*   wbase = wp + (size_t)(n0 + wrow) * Kh + wj4;

    v8f acc[4][2];
    #pragma unroll
    for (int i = 0; i < 4; ++i)
        #pragma unroll
        for (int j = 0; j < 2; ++j)
            acc[i][j] = 0.0f;

    const int nk = K / BK;

    // --- register staging buffers (one K-tile in flight) ---
    float4 xv[8];
    int4   wv[4];

    // prologue: issue all loads for tile 0 (no waits between them)
    #pragma unroll
    for (int i = 0; i < 8; ++i)
        xv[i] = *(const float4*)(xbase + (size_t)(16 * i) * K);
    #pragma unroll
    for (int i = 0; i < 4; ++i)
        wv[i] = *(const int4*)(wbase + (size_t)(32 * i) * Kh);

    for (int kt = 0; kt < nk; ++kt) {
        // ---- drain staged registers into LDS (single batched loadcnt wait here)
        #pragma unroll
        for (int i = 0; i < 8; ++i) {
            v4h h;
            h.x = (_Float16)xv[i].x; h.y = (_Float16)xv[i].y;
            h.z = (_Float16)xv[i].z; h.w = (_Float16)xv[i].w;
            *(v4h*)&As[xrow + 16 * i][xcol] = h;
        }
        #pragma unroll
        for (int i = 0; i < 4; ++i) {
            const int pk[4] = {wv[i].x, wv[i].y, wv[i].z, wv[i].w};
            #pragma unroll
            for (int c = 0; c < 4; ++c) {
                const int p = pk[c];
                v2h w2;
                w2.x = (_Float16)(float)((p << 28) >> 28);  // even k, sext int4
                w2.y = (_Float16)(float)((p << 24) >> 28);  // odd  k, sext int4
                *(v2h*)&Bs[wrow + 32 * i][wj4 * 2 + c * 2] = w2;
            }
        }
        __syncthreads();

        // ---- issue next tile's global loads; latency covered by WMMA below
        if (kt + 1 < nk) {
            const float* xp = xbase + (size_t)(kt + 1) * BK;
            const int*   wq = wbase + (size_t)(kt + 1) * (BK / 2);
            #pragma unroll
            for (int i = 0; i < 8; ++i)
                xv[i] = *(const float4*)(xp + (size_t)(16 * i) * K);
            #pragma unroll
            for (int i = 0; i < 4; ++i)
                wv[i] = *(const int4*)(wq + (size_t)(32 * i) * Kh);
        }

        // ---- compute: two K=32 WMMA steps, 8 WMMAs each
        #pragma unroll
        for (int ks = 0; ks < 2; ++ks) {
            const int kb = ks * 32;

            // A fragments (16x32 f16): lanes 0-15 row m, K {0..7,16..23};
            //                          lanes 16-31 same rows, K {8..15,24..31}
            const int arow = lane & 15;
            const int ka   = (lane >= 16) ? 8 : 0;
            v16h a[4];
            #pragma unroll
            for (int i = 0; i < 4; ++i) {
                const int row = wm + i * 16 + arow;
                v8h lo = *(const v8h*)&As[row][kb + ka];
                v8h hi = *(const v8h*)&As[row][kb + ka + 16];
                a[i] = __builtin_shufflevector(lo, hi,
                        0,1,2,3,4,5,6,7,8,9,10,11,12,13,14,15);
            }

            // B fragments (32x16 f16): lanes 0-15 col n, K {0..7,16..23};
            //                          lanes 16-31 same cols, K {8..15,24..31}
            const int bcol = lane & 15;
            const int kbo  = (lane >= 16) ? 8 : 0;
            v16h b[2];
            #pragma unroll
            for (int j = 0; j < 2; ++j) {
                const int col = wn + j * 16 + bcol;
                v8h lo = *(const v8h*)&Bs[col][kb + kbo];
                v8h hi = *(const v8h*)&Bs[col][kb + kbo + 16];
                b[j] = __builtin_shufflevector(lo, hi,
                        0,1,2,3,4,5,6,7,8,9,10,11,12,13,14,15);
            }

            #pragma unroll
            for (int i = 0; i < 4; ++i)
                #pragma unroll
                for (int j = 0; j < 2; ++j)
                    acc[i][j] = __builtin_amdgcn_wmma_f32_16x16x32_f16(
                        /*neg_a=*/false, a[i], /*neg_b=*/false, b[j],
                        /*c_mod=*/(short)0, acc[i][j],
                        /*reuse_a=*/false, /*reuse_b=*/false);
        }

        __syncthreads();
    }

    // ---- epilogue: out = acc * scale[n] + bias[n]
    // C layout: lane 0-15 -> N=lane, M=vgpr r; lane 16-31 -> N=lane-16, M=8+r
    const int ncol = lane & 15;
    const int rofs = (lane >= 16) ? 8 : 0;
    #pragma unroll
    for (int j = 0; j < 2; ++j) {
        const int ng = n0 + wn + j * 16 + ncol;
        const float s  = scale[ng];
        const float bb = bias[ng];
        #pragma unroll
        for (int i = 0; i < 4; ++i) {
            const int mg = m0 + wm + i * 16 + rofs;
            #pragma unroll
            for (int r = 0; r < 8; ++r) {
                out[(size_t)(mg + r) * N + ng] = acc[i][j][r] * s + bb;
            }
        }
    }
}

extern "C" void kernel_launch(void* const* d_in, const int* in_sizes, int n_in,
                              void* d_out, int out_size, void* d_ws, size_t ws_size,
                              hipStream_t stream) {
    const float* x     = (const float*)d_in[0];   // [B,S,IN] fp32
    const int*   wp    = (const int*)  d_in[1];   // [OUT, IN/2] int32 (uint8 payload)
    const float* scale = (const float*)d_in[2];   // [OUT,1]
    const float* bias  = (const float*)d_in[3];   // [OUT]
    float*       out   = (float*)d_out;           // [B,S,OUT] fp32

    const int N  = in_sizes[3];            // OUT (bias length)
    const int Kh = in_sizes[1] / N;        // IN/2
    const int K  = Kh * 2;                 // IN
    const int M  = in_sizes[0] / K;        // B*S

    // Dims (M=4096, N=11008, K=4096) divide tiles exactly: 86 x 32 blocks.
    dim3 grid(N / BN, M / BM);
    int4linear_wmma_f16<<<grid, 256, 0, stream>>>(x, wp, scale, bias, out, M, N, K, Kh);
}
